// LigerFusedLinearGRPOLoss_25941602467893
// MI455X (gfx1250) — matrix operand order repaired and checked by reading, not compiled, not run on hardware
//
#include <hip/hip_runtime.h>

#define HDIM 2048
#define LDA  (HDIM + 8)   // padded LDS row stride (halves) -> conflict-free ds_load_b128
#define NROWS 2048        // B*T
#define VDIM 32000
#define TLEN 512
#define MB 32             // rows per workgroup (2 M-tiles of 16)
#define VSPLIT 4
#define NWAVES 8
#define VTILES (VDIM / 16)        // 2000
#define VT_PER (VTILES / VSPLIT)  // 500
#define NEG_BIG (-3.0e38f)

typedef __attribute__((ext_vector_type(16))) __bf16        v16bf;
typedef __attribute__((ext_vector_type(8)))  float         v8f;
typedef __attribute__((ext_vector_type(4)))  unsigned int  u32x4;
typedef __attribute__((ext_vector_type(4)))  float         f32x4;
typedef __attribute__((ext_vector_type(4)))  unsigned short usv4;

struct U32x8 { u32x4 lo, hi; };

static __device__ __forceinline__ unsigned short f2bf(float f) {
  unsigned u = __float_as_uint(f);
  u += 0x7FFFu + ((u >> 16) & 1u);   // round-to-nearest-even
  return (unsigned short)(u >> 16);
}

static __device__ __forceinline__ v16bf frag2(u32x4 lo, u32x4 hi) {
  U32x8 u; u.lo = lo; u.hi = hi;
  return __builtin_bit_cast(v16bf, u);
}

// ---------------------------------------------------------------------------
// fp32 -> bf16 streaming conversion (one HBM pass; bf16 weights then live in L2)
// ---------------------------------------------------------------------------
__global__ void cvt_bf16_kernel(const float* __restrict__ src,
                                unsigned short* __restrict__ dst,
                                long long n4) {
  long long i = (long long)blockIdx.x * blockDim.x + threadIdx.x;
  long long stride = (long long)gridDim.x * blockDim.x;
  for (; i < n4; i += stride) {
    f32x4 v = reinterpret_cast<const f32x4*>(src)[i];
    usv4 r;
    r[0] = f2bf(v[0]); r[1] = f2bf(v[1]); r[2] = f2bf(v[2]); r[3] = f2bf(v[3]);
    reinterpret_cast<usv4*>(dst)[i] = r;
  }
}

// ---------------------------------------------------------------------------
// Fused GEMM (bf16 WMMA) + bias + online log-softmax/argmax partials.
// grid = (NROWS/MB, VSPLIT), block = 256 (8 waves).
// Each wave owns two 16x16 C tiles per vocab tile, sharing every B fragment.
// kdim is a RUNTIME argument so the K loop cannot be fully unrolled (which
// previously caused LICM to hoist all A fragments and spill them to scratch).
// B fragments are prefetched one K-step ahead via a wrap-around index.
// ---------------------------------------------------------------------------
__global__ void __launch_bounds__(256, 1)
gemm_lse_kernel(const float* __restrict__ x,              // [NROWS, HDIM] fp32
                const unsigned short* __restrict__ wbf,   // [VDIM, HDIM] bf16
                const float* __restrict__ bias,           // [VDIM]
                const int* __restrict__ chosen,           // null => policy pass
                float* __restrict__ pm, float* __restrict__ ps,
                int* __restrict__ pidx, float* __restrict__ pcap,
                const int kdim) {
  extern __shared__ char smem_raw[];
  unsigned short* ldsA = reinterpret_cast<unsigned short*>(smem_raw); // MB*LDA bf16
  float* red_m = reinterpret_cast<float*>(smem_raw + MB * LDA * 2);
  float* red_s = red_m + NWAVES * MB;
  float* red_c = red_s + NWAVES * MB;
  int*   red_i = reinterpret_cast<int*>(red_c + NWAVES * MB);

  const int tid  = threadIdx.x;
  const int wave = tid >> 5;
  const int lane = tid & 31;
  const int half = lane >> 4;
  const int l16  = lane & 15;
  const int m0   = blockIdx.x * MB;

  // Stage A rows into LDS as bf16 (fp32 global -> cvt -> ds_store, padded rows)
  {
    const float* src = x + (size_t)m0 * HDIM;
    for (int i = tid * 4; i < MB * HDIM; i += 256 * 4) {
      f32x4 v = *reinterpret_cast<const f32x4*>(src + i);
      usv4 r;
      r[0] = f2bf(v[0]); r[1] = f2bf(v[1]); r[2] = f2bf(v[2]); r[3] = f2bf(v[3]);
      int row = i >> 11;            // i / HDIM
      int c0  = i & (HDIM - 1);
      *reinterpret_cast<usv4*>(ldsA + row * LDA + c0) = r;
    }
  }
  __syncthreads();

  const bool is_ref = (chosen != nullptr);

  float sm[16], ss[16], sc[16];
  int   si[16], sch[16];
#pragma unroll
  for (int t = 0; t < 16; ++t) {
    sm[t] = NEG_BIG; ss[t] = 0.f; sc[t] = 0.f; si[t] = 0x7fffffff; sch[t] = -1;
  }
  if (is_ref) {
#pragma unroll
    for (int t = 0; t < 16; ++t) {
      int row = m0 + ((t >> 3) << 4) + (half << 3) + (t & 7);
      sch[t] = chosen[row];
    }
  }

  // A fragment bases (A order used here, matching B: row = lane % 16, 16 contiguous K per lane-half)
  const unsigned short* a0base = ldsA + l16 * LDA + 16 * half;
  const unsigned short* a1base = ldsA + (16 + l16) * LDA + 16 * half;

  const int jlo = blockIdx.y * VT_PER;
  for (int j = jlo + wave; j < jlo + VT_PER; j += NWAVES) {
    const int col = (j << 4) + l16;
    const float bv = bias[col];
    // B fragment: lane = column, 16 contiguous K per lane-half
    const unsigned short* brow = wbf + (size_t)col * HDIM + 16 * half;

    v8f acc0 = {}; v8f acc1 = {};
    v16bf bcur = *reinterpret_cast<const v16bf*>(brow);   // prefetch k=0
#pragma unroll 4
    for (int k = 0; k < kdim; k += 32) {
      const int kn = (k + 32) & (HDIM - 1);   // last iter wraps to 0 (harmless)
      v16bf bnext = *reinterpret_cast<const v16bf*>(brow + kn);
      u32x4 alo0 = *reinterpret_cast<const u32x4*>(a0base + k);
      u32x4 ahi0 = *reinterpret_cast<const u32x4*>(a0base + k + 8);
      u32x4 alo1 = *reinterpret_cast<const u32x4*>(a1base + k);
      u32x4 ahi1 = *reinterpret_cast<const u32x4*>(a1base + k + 8);
      acc0 = __builtin_amdgcn_wmma_f32_16x16x32_bf16(false, frag2(alo0, ahi0),
                                                     false, bcur, (short)0,
                                                     acc0, false, false);
      acc1 = __builtin_amdgcn_wmma_f32_16x16x32_bf16(false, frag2(alo1, ahi1),
                                                     false, bcur, (short)0,
                                                     acc1, false, false);
      bcur = bnext;
    }

    // Online softmax / argmax / capture epilogue (per-lane, no shuffles here)
#pragma unroll
    for (int r = 0; r < 8; ++r) {
      {
        float v_ = acc0[r] + bv;
        float om = sm[r];
        float nm = fmaxf(om, v_);
        ss[r] = ss[r] * __expf(om - nm) + __expf(v_ - nm);
        if (v_ > om) si[r] = col;
        sm[r] = nm;
        if (is_ref && col == sch[r]) sc[r] = v_;
      }
      {
        int t = 8 + r;
        float v_ = acc1[r] + bv;
        float om = sm[t];
        float nm = fmaxf(om, v_);
        ss[t] = ss[t] * __expf(om - nm) + __expf(v_ - nm);
        if (v_ > om) si[t] = col;
        sm[t] = nm;
        if (is_ref && col == sch[t]) sc[t] = v_;
      }
    }
  }

  // Butterfly merge within each 16-lane half (halves hold disjoint row sets)
#pragma unroll
  for (int t = 0; t < 16; ++t) {
#pragma unroll
    for (int off = 1; off < 16; off <<= 1) {
      float om = __shfl_xor(sm[t], off, 32);
      float os = __shfl_xor(ss[t], off, 32);
      float oc = __shfl_xor(sc[t], off, 32);
      int   oi = __shfl_xor(si[t], off, 32);
      float nm = fmaxf(sm[t], om);
      ss[t] = ss[t] * __expf(sm[t] - nm) + os * __expf(om - nm);
      if (om > sm[t] || (om == sm[t] && oi < si[t])) si[t] = oi;
      sm[t] = nm;
      sc[t] += oc;
    }
  }

  if (l16 == 0) {
#pragma unroll
    for (int t = 0; t < 16; ++t) {
      int lr = ((t >> 3) << 4) + (half << 3) + (t & 7);
      red_m[wave * MB + lr] = sm[t];
      red_s[wave * MB + lr] = ss[t];
      red_c[wave * MB + lr] = sc[t];
      red_i[wave * MB + lr] = si[t];
    }
  }
  __syncthreads();

  // Cross-wave merge; one thread per local row writes the V-split partial
  if (tid < MB) {
    float m = NEG_BIG, s = 0.f, c = 0.f;
    int idx = 0x7fffffff;
    for (int w = 0; w < NWAVES; ++w) {
      float om = red_m[w * MB + tid];
      float os = red_s[w * MB + tid];
      int   oi = red_i[w * MB + tid];
      float nm = fmaxf(m, om);
      s = s * __expf(m - nm) + os * __expf(om - nm);
      if (om > m || (om == m && oi < idx)) idx = oi;
      m = nm;
      c += red_c[w * MB + tid];
    }
    int o = (m0 + tid) * VSPLIT + blockIdx.y;
    pm[o] = m; ps[o] = s; pidx[o] = idx; pcap[o] = c;
  }
}

// ---------------------------------------------------------------------------
// Merge V-split partials per row. Policy: chosen + tok_lp = -log(S).
// Ref: ref_tok_lp = captured_logit[chosen] - (M + log S).
// ---------------------------------------------------------------------------
__global__ void finalize_kernel(const float* __restrict__ pm,
                                const float* __restrict__ ps,
                                const int* __restrict__ pidx,
                                const float* __restrict__ pcap,
                                int* __restrict__ chosen_out,
                                float* __restrict__ lp_out, int is_ref) {
  int row = blockIdx.x * blockDim.x + threadIdx.x;
  if (row >= NROWS) return;
  float m = NEG_BIG, s = 0.f, c = 0.f;
  int idx = 0x7fffffff;
  for (int p = 0; p < VSPLIT; ++p) {
    int o = row * VSPLIT + p;
    float om = pm[o], os = ps[o];
    float nm = fmaxf(m, om);
    s = s * __expf(m - nm) + os * __expf(om - nm);
    int oi = pidx[o];
    if (om > m || (om == m && oi < idx)) idx = oi;
    m = nm;
    c += pcap[o];
  }
  if (is_ref) {
    lp_out[row] = c - (m + logf(s));     // ref_tok_lp at policy's argmax
  } else {
    chosen_out[row] = idx;
    lp_out[row] = -logf(s);              // tok_lp = max - lse = -log(S)
  }
}

// ---------------------------------------------------------------------------
// Final GRPO loss: advantages (ddof=1), ratio==1 forward, KL, masked mean.
// ---------------------------------------------------------------------------
__global__ void loss_kernel(const float* __restrict__ lp,
                            const float* __restrict__ ref_lp,
                            const float* __restrict__ mask,
                            const float* __restrict__ rewards,
                            float* __restrict__ out) {
  __shared__ float redL[8], redM[8];
  int tid = threadIdx.x;
  float rmean = 0.f;
  for (int b = 0; b < 4; ++b) rmean += rewards[b];
  rmean *= 0.25f;
  float var = 0.f;
  for (int b = 0; b < 4; ++b) { float d = rewards[b] - rmean; var += d * d; }
  float sd = sqrtf(var / 3.0f);          // ddof = 1

  float ls = 0.f, ms = 0.f;
  for (int i = tid; i < NROWS; i += 256) {
    int b = i >> 9;                      // i / TLEN
    float adv = (rewards[b] - rmean) / (sd + 1e-4f);
    float a = mask[i];
    float d = ref_lp[i] - lp[i];
    float kl = __expf(d) - d - 1.0f;
    ls += (-adv + 0.1f * kl) * a;        // ratio == 1 in fwd
    ms += a;
  }
  for (int off = 16; off > 0; off >>= 1) {
    ls += __shfl_down(ls, off, 32);
    ms += __shfl_down(ms, off, 32);
  }
  if ((tid & 31) == 0) { redL[tid >> 5] = ls; redM[tid >> 5] = ms; }
  __syncthreads();
  if (tid == 0) {
    float L = 0.f, M = 0.f;
    for (int w = 0; w < 8; ++w) { L += redL[w]; M += redM[w]; }
    out[0] = L / fmaxf(M, 1.0f);
  }
}

// ---------------------------------------------------------------------------
extern "C" void kernel_launch(void* const* d_in, const int* in_sizes, int n_in,
                              void* d_out, int out_size, void* d_ws, size_t ws_size,
                              hipStream_t stream) {
  const float* x       = (const float*)d_in[0];
  const float* weight  = (const float*)d_in[1];
  const float* mask    = (const float*)d_in[2];
  const float* rewards = (const float*)d_in[3];
  const float* bias    = (const float*)d_in[4];
  const float* rx      = (const float*)d_in[5];
  const float* rweight = (const float*)d_in[6];
  const float* rbias   = (const float*)d_in[7];

  char* ws = (char*)d_ws;
  const size_t WB = (size_t)VDIM * HDIM * sizeof(unsigned short); // 131,072,000 B
  unsigned short* wbf  = (unsigned short*)(ws);
  unsigned short* rwbf = (unsigned short*)(ws + WB);
  char* p = ws + 2 * WB;
  float* pm   = (float*)p;  p += (size_t)NROWS * VSPLIT * 4;
  float* ps   = (float*)p;  p += (size_t)NROWS * VSPLIT * 4;
  int*   pidx = (int*)p;    p += (size_t)NROWS * VSPLIT * 4;
  float* pcap = (float*)p;  p += (size_t)NROWS * VSPLIT * 4;
  int*   chosen = (int*)p;  p += (size_t)NROWS * 4;
  float* lp     = (float*)p; p += (size_t)NROWS * 4;
  float* rlp    = (float*)p; p += (size_t)NROWS * 4;

  const long long n4 = (long long)VDIM * HDIM / 4;
  cvt_bf16_kernel<<<4096, 256, 0, stream>>>(weight,  wbf,  n4);
  cvt_bf16_kernel<<<4096, 256, 0, stream>>>(rweight, rwbf, n4);

  const size_t smem = (size_t)MB * LDA * 2 + (size_t)4 * NWAVES * MB * 4;
  dim3 ggrid(NROWS / MB, VSPLIT);

  // Policy pass: argmax + tok_lp
  gemm_lse_kernel<<<ggrid, 256, smem, stream>>>(x, wbf, bias, nullptr,
                                                pm, ps, pidx, pcap, HDIM);
  finalize_kernel<<<NROWS / 256, 256, 0, stream>>>(pm, ps, pidx, pcap,
                                                   chosen, lp, 0);
  // Ref pass: lse + gather at policy's argmax
  gemm_lse_kernel<<<ggrid, 256, smem, stream>>>(rx, rwbf, rbias, chosen,
                                                pm, ps, pidx, pcap, HDIM);
  finalize_kernel<<<NROWS / 256, 256, 0, stream>>>(pm, ps, pidx, pcap,
                                                   chosen, rlp, 1);

  loss_kernel<<<1, 256, 0, stream>>>(lp, rlp, mask, rewards, (float*)d_out);
}
